// GATModel_67242007986930
// MI455X (gfx1250) — matrix operand order repaired and checked by reading, not compiled, not run on hardware
//
#include <hip/hip_runtime.h>

// ---------------------------------------------------------------------------
// HeteroGAT (2 layers, 8 relations, 5 node types) for MI455X / gfx1250.
// GEMMs run in bf16 on v_wmma_f32_16x16x32_bf16 (2 independent accumulator
// chains per wave to hide WMMA RAW hazards); edge softmax/aggregation via
// L2-resident atomics (hs/hd, alpha and per-type accumulators all fit in the
// 192MB L2).
// ---------------------------------------------------------------------------

#define NNODE 100000
#define NEDGE 300000
#define DDIM  128      // H*C = 4*32, also IN for layer 1
#define KDIM  128
#define HHEAD 4
#define NREL  8
#define LDSK  130      // padded LDS row stride (ushorts) to avoid bank conflicts

typedef unsigned short u16;
typedef __attribute__((ext_vector_type(16))) __bf16 v16bf;
typedef __attribute__((ext_vector_type(16))) unsigned short v16u;
typedef __attribute__((ext_vector_type(8)))  float  v8f;

// ----------------------------- helpers -------------------------------------

__device__ __forceinline__ u16 f2bf(float f) {
  unsigned u = __float_as_uint(f);
  unsigned r = u + 0x7fffu + ((u >> 16) & 1u);   // round-to-nearest-even
  if ((u & 0x7f800000u) == 0x7f800000u) r = u;   // inf/nan pass-through
  return (u16)(r >> 16);
}

// order-preserving encoding of float into uint so atomicMax works for floats
__device__ __forceinline__ unsigned fenc(float f) {
  unsigned u = __float_as_uint(f);
  return (u & 0x80000000u) ? ~u : (u | 0x80000000u);
}
__device__ __forceinline__ float fdec(unsigned e) {
  unsigned u = (e & 0x80000000u) ? (e & 0x7fffffffu) : ~e;
  return __uint_as_float(u);
}

// ----------------------------- small kernels --------------------------------

__global__ void fill_f32(float* __restrict__ p, float v, long n) {
  long i = (long)blockIdx.x * blockDim.x + threadIdx.x;
  if (i < n) p[i] = v;
}
__global__ void fill_u32(unsigned* __restrict__ p, unsigned v, long n) {
  long i = (long)blockIdx.x * blockDim.x + threadIdx.x;
  if (i < n) p[i] = v;
}
__global__ void cvt_f32_bf16(const float* __restrict__ s, u16* __restrict__ d, long n) {
  long i = (long)blockIdx.x * blockDim.x + threadIdx.x;
  if (i < n) d[i] = f2bf(s[i]);
}

// ----------------------------- WMMA GEMM ------------------------------------
// C[M x 128] = A[M x 128] * B[128 x 128], A/B bf16 (as u16 bits), C f32.
// One block = 256 threads = 8 waves; block tile = 32 rows x 128 cols.
// Each wave owns a 16-col slab and TWO 16x16 accumulators (rows [0,16) and
// [16,32) of the tile) so the two WMMA chains are independent and the
// scheduler can interleave them across the bf16 WMMA RAW hazard window.
// B is staged transposed in LDS (Bt[n][k], padded stride) so each lane's
// 16-element K-slice is contiguous and bank-conflict-free.
__global__ __launch_bounds__(256)
void gemm_bf16_wmma(const u16* __restrict__ A, const u16* __restrict__ B,
                    float* __restrict__ C, int M) {
  __shared__ u16 Bt[DDIM * LDSK];   // transposed B, padded  (~32.5 KB)
  __shared__ u16 At[32   * LDSK];   // A tile, padded        (~8.1 KB)
  const int tid = threadIdx.x;
  const int m0  = blockIdx.x * 32;

  // cooperative load of B (row-major in memory), store transposed
  for (int i = tid; i < KDIM * DDIM; i += 256) {
    int k = i >> 7, n = i & 127;            // coalesced read of B[k][n]
    Bt[n * LDSK + k] = B[i];
  }
  // cooperative load of 32x128 A tile (contiguous rows)
  for (int i = tid; i < 32 * KDIM; i += 256) {
    int r = i >> 7, k = i & 127;
    At[r * LDSK + k] = A[(long)(m0 + r) * KDIM + k];
  }
  __syncthreads();

  const int wid  = tid >> 5;        // wave id 0..7 -> 16-col slab
  const int lane = tid & 31;
  const int half = lane >> 4;       // 0: lanes 0-15, 1: lanes 16-31
  const int l15  = lane & 15;
  const int n0   = wid * 16;

  v8f acc0 = {0.f, 0.f, 0.f, 0.f, 0.f, 0.f, 0.f, 0.f};
  v8f acc1 = {0.f, 0.f, 0.f, 0.f, 0.f, 0.f, 0.f, 0.f};

#pragma unroll
  for (int kk = 0; kk < KDIM; kk += 32) {
    v16u a0u, a1u, bu;
    // A frag (16x32 bf16): lanes 0-15 hold row l15, K = kk+{0..7,16..23};
    //                      lanes 16-31 hold row l15, K = kk+{8..15,24..31}.
    const u16* ar0 = &At[l15 * LDSK + kk + half * 8];          // rows 0..15
    const u16* ar1 = &At[(16 + l15) * LDSK + kk + half * 8];   // rows 16..31
#pragma unroll
    for (int j = 0; j < 8; ++j) {
      a0u[j] = ar0[j]; a0u[j + 8] = ar0[16 + j];
      a1u[j] = ar1[j]; a1u[j + 8] = ar1[16 + j];
    }
    // B frag (32x16 bf16): lanes 0-15 hold col n0+l15, K = kk+{0..7,16..23};
    //                      lanes 16-31 hold col n0+l15, K = kk+{8..15,24..31}.
    const u16* bc = &Bt[(n0 + l15) * LDSK + kk + half * 8];
#pragma unroll
    for (int j = 0; j < 16; ++j) bu[j] = bc[j < 8 ? j : 8 + j];

    v16bf a0 = __builtin_bit_cast(v16bf, a0u);
    v16bf a1 = __builtin_bit_cast(v16bf, a1u);
    v16bf b  = __builtin_bit_cast(v16bf, bu);
    acc0 = __builtin_amdgcn_wmma_f32_16x16x32_bf16(
        false, a0, false, b, (short)0, acc0, false, false);
    acc1 = __builtin_amdgcn_wmma_f32_16x16x32_bf16(
        false, a1, false, b, (short)0, acc1, false, false);
  }

  // C layout: VGPR i -> M = i + 8*half, N = l15 (per ISA 16x16 f32 map)
#pragma unroll
  for (int i = 0; i < 8; ++i) {
    int row = m0 + i + half * 8;
    C[(long)row * DDIM + n0 + l15]        = acc0[i];
    C[(long)(row + 16) * DDIM + n0 + l15] = acc1[i];
  }
}

// ---------------------- per-node head logit terms ---------------------------
// es[n,h] = sum_c h[n, h*32+c] * a[h, c]
__global__ void head_logits(const float* __restrict__ h, const float* __restrict__ a,
                            float* __restrict__ out, long n4) {
  long idx = (long)blockIdx.x * blockDim.x + threadIdx.x;
  if (idx >= n4) return;
  long n = idx >> 2;
  int  hh = (int)(idx & 3);
  const float* hp = h + n * DDIM + hh * 32;
  const float* ap = a + hh * 32;
  float s = 0.f;
#pragma unroll
  for (int c = 0; c < 32; ++c) s += hp[c] * ap[c];
  out[idx] = s;
}

// ----------------------------- edge passes ----------------------------------

// pass 1: per-(edge,head) leaky-relu logit -> atomic segment max over dst
__global__ void edge_max(const int* __restrict__ src, const int* __restrict__ dst,
                         const float* __restrict__ es, const float* __restrict__ ed,
                         unsigned* __restrict__ menc, long e4) {
  long idx = (long)blockIdx.x * blockDim.x + threadIdx.x;
  if (idx >= e4) return;
  long e = idx >> 2; int h = (int)(idx & 3);
  int s = src[e], d = dst[e];
  float x = es[(long)s * 4 + h] + ed[(long)d * 4 + h];
  float l = x > 0.f ? x : 0.2f * x;
  atomicMax(&menc[(long)d * 4 + h], fenc(l));
}

// pass 2: p = exp(logit - m); stash p per (edge,head), atomic segment sum
__global__ void edge_p(const int* __restrict__ src, const int* __restrict__ dst,
                       const float* __restrict__ es, const float* __restrict__ ed,
                       const unsigned* __restrict__ menc, float* __restrict__ palpha,
                       float* __restrict__ denom, long e4) {
  long idx = (long)blockIdx.x * blockDim.x + threadIdx.x;
  if (idx >= e4) return;
  long e = idx >> 2; int h = (int)(idx & 3);
  int s = src[e], d = dst[e];
  float x = es[(long)s * 4 + h] + ed[(long)d * 4 + h];
  float l = x > 0.f ? x : 0.2f * x;
  unsigned me = menc[(long)d * 4 + h];
  float m = me ? fdec(me) : 0.f;
  float p = __expf(l - m);
  palpha[idx] = p;
  atomicAdd(&denom[(long)d * 4 + h], p);
}

// pass 3: normalize p -> alpha in place
__global__ void edge_alpha(const int* __restrict__ dst, const float* __restrict__ denom,
                           float* __restrict__ palpha, long e4) {
  long idx = (long)blockIdx.x * blockDim.x + threadIdx.x;
  if (idx >= e4) return;
  long e = idx >> 2; int h = (int)(idx & 3);
  int d = dst[e];
  palpha[idx] = palpha[idx] / (denom[(long)d * 4 + h] + 1e-16f);
}

// pass 4: weighted scatter-add. One thread per (edge, channel); 128
// consecutive threads share an edge so atomic adds to out[dst] are contiguous
// and hs reads are fully coalesced. alpha/index loads hit L1/L2.
__global__ void edge_agg(const int* __restrict__ src, const int* __restrict__ dst,
                         const float* __restrict__ palpha, const float* __restrict__ hs,
                         float* __restrict__ out, long total) {
  long idx = (long)blockIdx.x * blockDim.x + threadIdx.x;
  if (idx >= total) return;
  long e = idx >> 7;
  int  c = (int)(idx & 127);
  int  h = c >> 5;
  int s = src[e], d = dst[e];
  float alpha = palpha[e * 4 + h];
  atomicAdd(&out[(long)d * DDIM + c], hs[(long)s * DDIM + c] * alpha);
}

// ----------------------------- finalize -------------------------------------

// bsum[t][c] = sum of b[r][c] over relations whose destination is type t
__global__ void bias_sums(const float* __restrict__ b, float* __restrict__ bsum) {
  int t = blockIdx.x;     // 0..4
  int c = threadIdx.x;    // 0..127
  float v;
  if (t == 0) v = b[1 * DDIM + c] + b[3 * DDIM + c] + b[5 * DDIM + c] + b[7 * DDIM + c];
  else        v = b[(2 * (t - 1)) * DDIM + c];
  bsum[t * DDIM + c] = v;
}

// layer-1 finalize: y = elu(acc + bias); requantize to bf16 for next layer,
// and zero the accumulator for layer-2 reuse.
__global__ void finalize_elu_bf16(float* __restrict__ acc, const float* __restrict__ bsum,
                                  u16* __restrict__ xb, long n) {
  long i = (long)blockIdx.x * blockDim.x + threadIdx.x;
  if (i >= n) return;
  int c = (int)(i & 127);
  float v = acc[i] + bsum[c];
  float y = v > 0.f ? v : (__expf(v) - 1.0f);
  xb[i] = f2bf(y);
  acc[i] = 0.f;
}

// layer-2 finalize: y = elu(acc + bias) -> f32 output
__global__ void finalize_elu_f32(const float* __restrict__ acc, const float* __restrict__ bsum,
                                 float* __restrict__ out, long n) {
  long i = (long)blockIdx.x * blockDim.x + threadIdx.x;
  if (i >= n) return;
  int c = (int)(i & 127);
  float v = acc[i] + bsum[c];
  out[i] = v > 0.f ? v : (__expf(v) - 1.0f);
}

// ----------------------------- driver ---------------------------------------

extern "C" void kernel_launch(void* const* d_in, const int* in_sizes, int n_in,
                              void* d_out, int out_size, void* d_ws, size_t ws_size,
                              hipStream_t stream) {
  (void)in_sizes; (void)n_in; (void)out_size; (void)ws_size;

  const float* x_in[5] = {(const float*)d_in[0], (const float*)d_in[1],
                          (const float*)d_in[2], (const float*)d_in[3],
                          (const float*)d_in[4]};
  const int*   edges = (const int*)d_in[5];
  const float* Ws[2] = {(const float*)d_in[6],  (const float*)d_in[11]};
  const float* Wd[2] = {(const float*)d_in[7],  (const float*)d_in[12]};
  const float* As[2] = {(const float*)d_in[8],  (const float*)d_in[13]};
  const float* Ad[2] = {(const float*)d_in[9],  (const float*)d_in[14]};
  const float* Bb[2] = {(const float*)d_in[10], (const float*)d_in[15]};
  float* outp = (float*)d_out;

  // ---- workspace carve-out --------------------------------------------------
  char*  ws  = (char*)d_ws;
  size_t off = 0;
  auto alloc = [&](size_t bytes) -> void* {
    void* p = ws + off;
    off = (off + bytes + 255) & ~(size_t)255;
    return p;
  };
  const long ND = (long)NNODE * DDIM;            // 12.8M
  u16* xb[5];
  for (int t = 0; t < 5; ++t) xb[t] = (u16*)alloc(ND * sizeof(u16));
  u16* Wsb[2] = {(u16*)alloc((size_t)NREL * KDIM * DDIM * 2),
                 (u16*)alloc((size_t)NREL * KDIM * DDIM * 2)};
  u16* Wdb[2] = {(u16*)alloc((size_t)NREL * KDIM * DDIM * 2),
                 (u16*)alloc((size_t)NREL * KDIM * DDIM * 2)};
  float*    hs     = (float*)alloc(ND * sizeof(float));
  float*    hd     = (float*)alloc(ND * sizeof(float));
  float*    es     = (float*)alloc((long)NNODE * HHEAD * sizeof(float));
  float*    ed     = (float*)alloc((long)NNODE * HHEAD * sizeof(float));
  unsigned* menc   = (unsigned*)alloc((long)NNODE * HHEAD * sizeof(unsigned));
  float*    denom  = (float*)alloc((long)NNODE * HHEAD * sizeof(float));
  float*    palpha = (float*)alloc((long)NEDGE * HHEAD * sizeof(float));
  float*    acc[5];
  for (int t = 0; t < 5; ++t) acc[t] = (float*)alloc(ND * sizeof(float));
  float*    bsum   = (float*)alloc(5 * DDIM * sizeof(float));

  auto cdiv = [](long a, long b) -> unsigned { return (unsigned)((a + b - 1) / b); };

  // ---- conversions + accumulator init (deterministic, every call) -----------
  for (int t = 0; t < 5; ++t)
    cvt_f32_bf16<<<cdiv(ND, 256), 256, 0, stream>>>(x_in[t], xb[t], ND);
  const long WN = (long)NREL * KDIM * DDIM;
  for (int l = 0; l < 2; ++l) {
    cvt_f32_bf16<<<cdiv(WN, 256), 256, 0, stream>>>(Ws[l], Wsb[l], WN);
    cvt_f32_bf16<<<cdiv(WN, 256), 256, 0, stream>>>(Wd[l], Wdb[l], WN);
  }
  for (int t = 0; t < 5; ++t)
    fill_f32<<<cdiv(ND, 256), 256, 0, stream>>>(acc[t], 0.f, ND);

  const int REL_S[NREL] = {0, 1, 0, 2, 0, 3, 0, 4};
  const int REL_D[NREL] = {1, 0, 2, 0, 3, 0, 4, 0};

  const long n4   = (long)NNODE * HHEAD;
  const long e4   = (long)NEDGE * HHEAD;
  const long etot = (long)NEDGE * DDIM;
  const unsigned gemmGrid = NNODE / 32;          // 3125 blocks of 256 threads

  for (int l = 0; l < 2; ++l) {
    for (int r = 0; r < NREL; ++r) {
      const int si = REL_S[r], di = REL_D[r];
      const int* srcp = edges + ((long)r * 2 + 0) * NEDGE;
      const int* dstp = edges + ((long)r * 2 + 1) * NEDGE;

      // projections on the matrix pipes
      gemm_bf16_wmma<<<gemmGrid, 256, 0, stream>>>(xb[si], Wsb[l] + (long)r * KDIM * DDIM, hs, NNODE);
      gemm_bf16_wmma<<<gemmGrid, 256, 0, stream>>>(xb[di], Wdb[l] + (long)r * KDIM * DDIM, hd, NNODE);

      // per-node logit terms
      head_logits<<<cdiv(n4, 256), 256, 0, stream>>>(hs, As[l] + (long)r * DDIM, es, n4);
      head_logits<<<cdiv(n4, 256), 256, 0, stream>>>(hd, Ad[l] + (long)r * DDIM, ed, n4);

      // segment softmax over incoming edges
      fill_u32<<<cdiv(n4, 256), 256, 0, stream>>>(menc, 0u, n4);
      fill_f32<<<cdiv(n4, 256), 256, 0, stream>>>(denom, 0.f, n4);
      edge_max<<<cdiv(e4, 256), 256, 0, stream>>>(srcp, dstp, es, ed, menc, e4);
      edge_p<<<cdiv(e4, 256), 256, 0, stream>>>(srcp, dstp, es, ed, menc, palpha, denom, e4);
      edge_alpha<<<cdiv(e4, 256), 256, 0, stream>>>(dstp, denom, palpha, e4);
      edge_agg<<<cdiv(etot, 256), 256, 0, stream>>>(srcp, dstp, palpha, hs, acc[di], etot);
    }

    bias_sums<<<5, DDIM, 0, stream>>>(Bb[l], bsum);
    for (int t = 0; t < 5; ++t) {
      if (l == 0) {
        finalize_elu_bf16<<<cdiv(ND, 256), 256, 0, stream>>>(acc[t], bsum + t * DDIM, xb[t], ND);
      } else {
        finalize_elu_f32<<<cdiv(ND, 256), 256, 0, stream>>>(acc[t], bsum + t * DDIM,
                                                            outp + (long)t * ND, ND);
      }
    }
  }
}
